// GaussianLearner_29368986370661
// MI455X (gfx1250) — matrix operand order repaired and checked by reading, not compiled, not run on hardware
//
#include <hip/hip_runtime.h>
#include <hip/hip_bf16.h>

typedef __attribute__((ext_vector_type(16))) _Float16 v16h;
typedef __attribute__((ext_vector_type(8)))  _Float16 v8h;
typedef __attribute__((ext_vector_type(2)))  _Float16 h2;
typedef __attribute__((ext_vector_type(8)))  float    v8f;
typedef float f2u __attribute__((ext_vector_type(2), aligned(4)));   // align-4 float2 load

#define C_CH   24
#define RES_0  200
#define RES_1  400
#define MLP_HW 168
#define OUT_CH 8
#define NPTS   1000000

#define K1P 64            // padded input feature dim (48 -> 64); col 48 = bias 1.0
#define NP  192           // padded hidden dim (168 -> 192); col 168 = bias 1.0
#define THREADS 384       // 12 waves (wave32)
#define WAVES (THREADS/32)
#define PTS_PER_BLOCK (WAVES*16)   // 192 points per block tile

// LDS carve-out (halfword offsets)
#define OFF_W1 0                         // 192*64  (w1[n][k], k=48 holds b1[n])
#define OFF_W2 (OFF_W1 + NP*K1P)         // 192*192 (w2[n][k], k=168 holds b2[n])
#define OFF_W3 (OFF_W2 + NP*NP)          // 16*192  (w3[n][k], k=168 holds b3[n])
#define OFF_FEAT (OFF_W3 + 16*NP)        // 192*64 halfwords
#define OFF_H1 (OFF_FEAT + PTS_PER_BLOCK*K1P)
#define OFF_H2 (OFF_H1 + PTS_PER_BLOCK*NP)
#define SMEM_HALF (OFF_H2 + PTS_PER_BLOCK*NP)
#define SMEM_BYTES (SMEM_HALF*2)         // = 276480 bytes < 320KB/WGP

// --- A fragment: 16x32 f16, rows (points), K-contiguous in LDS ---
__device__ __forceinline__ v16h load_a_frag(const _Float16* base, int strideH, int kt, int lane) {
  int m  = lane & 15;
  int kb = kt * 32 + ((lane & 16) ? 16 : 0);
  const _Float16* p = base + m * strideH + kb;
  v8h lo = *(const v8h*)(p);        // K = kb .. kb+7
  v8h hi = *(const v8h*)(p + 8);    // K = kb+8 .. kb+15
  return __builtin_shufflevector(lo, hi, 0,1,2,3,4,5,6,7,8,9,10,11,12,13,14,15);
}

// --- B fragment: 32x16 f16, weights stored [n][k] K-contiguous in LDS ---
__device__ __forceinline__ v16h load_b_frag(const _Float16* w, int strideH, int kt, int nt, int lane) {
  int n  = nt * 16 + (lane & 15);
  int kb = kt * 32 + ((lane & 16) ? 16 : 0);
  return *(const v16h*)(w + n * strideH + kb);   // 32B contiguous, aligned
}

// Relu + f32->f16 + store 8 accumulator rows (packed cvt, packed max, b16/d16_hi stores)
__device__ __forceinline__ void store_relu8(_Float16* dst, int n, int mOff, v8f acc) {
  const h2 z = {(_Float16)0.0f, (_Float16)0.0f};
  #pragma unroll
  for (int r = 0; r < 8; r += 2) {
    h2 p;
    p.x = (_Float16)acc[r];       // fuses to v_cvt_pk_f16_f32
    p.y = (_Float16)acc[r + 1];
    p = __builtin_elementwise_max(p, z);   // relu in packed f16 (cvt monotonic, 0 exact)
    dst[(mOff + r) * NP + n]     = p.x;
    dst[(mOff + r + 1) * NP + n] = p.y;
  }
}

// Bilinear grid_sample accumulate (align_corners=True), 24 channels.
// In-range coords -> clamp+shift is exact, no validity masks; x-taps as one float2.
__device__ __forceinline__ void gs_accum(const float* __restrict__ plane, int R,
                                         float gx, float gy, float* acc) {
  float xf = (gx + 1.0f) * 0.5f * (float)(R - 1);
  float yf = (gy + 1.0f) * 0.5f * (float)(R - 1);
  xf = fminf(fmaxf(xf, 0.0f), (float)(R - 1));
  yf = fminf(fmaxf(yf, 0.0f), (float)(R - 1));
  int x0 = (int)xf;  if (x0 > R - 2) x0 = R - 2;   // shift; weight absorbs it
  int y0 = (int)yf;  if (y0 > R - 2) y0 = R - 2;
  float wx = xf - (float)x0;
  float wy = yf - (float)y0;
  float w11 = wx * wy;
  float w01 = wx - w11;           // wx*(1-wy)
  float w10 = wy - w11;           // (1-wx)*wy
  float w00 = 1.0f - wx - w10;    // (1-wx)*(1-wy)
  const float* row0 = plane + (y0 * R + x0);
  const float* row1 = row0 + R;
  int chS = R * R;
  #pragma unroll
  for (int c = 0; c < C_CH; ++c) {
    f2u t0 = *(const f2u*)(row0 + c * chS);
    f2u t1 = *(const f2u*)(row1 + c * chS);
    float a = acc[c];
    a = fmaf(t0.x, w00, a);
    a = fmaf(t0.y, w01, a);
    a = fmaf(t1.x, w10, a);
    a = fmaf(t1.y, w11, a);
    acc[c] = a;
  }
}

extern "C" __global__ __launch_bounds__(THREADS, 1)
void triplane_mlp_wmma(const float* __restrict__ xyz,
                       const float* __restrict__ xy0, const float* __restrict__ xz0,
                       const float* __restrict__ yz0,
                       const float* __restrict__ xy1, const float* __restrict__ xz1,
                       const float* __restrict__ yz1,
                       const float* __restrict__ W1, const float* __restrict__ b1,
                       const float* __restrict__ W2, const float* __restrict__ b2,
                       const float* __restrict__ W3, const float* __restrict__ b3,
                       float* __restrict__ out, int nTiles) {
  extern __shared__ _Float16 smem[];
  _Float16* w1s   = smem + OFF_W1;
  _Float16* w2s   = smem + OFF_W2;
  _Float16* w3s   = smem + OFF_W3;
  _Float16* feats = smem + OFF_FEAT;
  _Float16* h1s   = smem + OFF_H1;
  _Float16* h2s   = smem + OFF_H2;

  const int tid = threadIdx.x;

  // ---- Stage weights (+ bias column) into LDS as f16, zero-padded ----
  for (int i = tid; i < NP * K1P; i += THREADS) {
    int n = i / K1P, k = i - n * K1P;
    float v = 0.0f;
    if (n < MLP_HW) {
      if (k < 2 * C_CH)       v = W1[n * (2 * C_CH) + k];
      else if (k == 2 * C_CH) v = b1[n];                  // bias column (K=48)
    }
    w1s[i] = (_Float16)v;
  }
  for (int i = tid; i < NP * NP; i += THREADS) {
    int n = i / NP, k = i - n * NP;
    float v = 0.0f;
    if (n < MLP_HW) {
      if (k < MLP_HW)       v = W2[n * MLP_HW + k];
      else if (k == MLP_HW) v = b2[n];                    // bias column (K=168)
    }
    w2s[i] = (_Float16)v;
  }
  for (int i = tid; i < 16 * NP; i += THREADS) {
    int n = i / NP, k = i - n * NP;
    float v = 0.0f;
    if (n < OUT_CH) {
      if (k < MLP_HW)       v = W3[n * MLP_HW + k];
      else if (k == MLP_HW) v = b3[n];                    // bias column (K=168)
    }
    w3s[i] = (_Float16)v;
  }
  // Feature buffer: pad zero, except col 48 = 1.0 (bias input); cols <48 rewritten each tile
  for (int i = tid; i < PTS_PER_BLOCK * K1P; i += THREADS) {
    int k = i & (K1P - 1);
    feats[i] = (_Float16)((k == 2 * C_CH) ? 1.0f : 0.0f);
  }
  __syncthreads();

  const int lane = tid & 31;
  const int wave = tid >> 5;
  const int m    = lane & 15;
  const int lvl  = lane >> 4;          // 0: res0 planes, 1: res1 planes
  const int mOff = lvl * 8;            // C/D row offset for this half-wave
  _Float16* featW = feats + wave * 16 * K1P;
  _Float16* h1W   = h1s   + wave * 16 * NP;
  _Float16* h2W   = h2s   + wave * 16 * NP;

  for (int tile = blockIdx.x; tile < nTiles; tile += gridDim.x) {
    const int ptBase = tile * PTS_PER_BLOCK + wave * 16;

    // ---- Phase 1: triplane bilinear features (wave-private) ----
    {
      int pt = ptBase + m;
      int pc = pt < NPTS ? pt : NPTS - 1;
      float px = __builtin_nontemporal_load(&xyz[pc * 3 + 0]);
      float py = __builtin_nontemporal_load(&xyz[pc * 3 + 1]);
      float pz = __builtin_nontemporal_load(&xyz[pc * 3 + 2]);
      const float* pXY = lvl ? xy1 : xy0;
      const float* pXZ = lvl ? xz1 : xz0;
      const float* pYZ = lvl ? yz1 : yz0;
      int R = lvl ? RES_1 : RES_0;
      float acc[C_CH];
      #pragma unroll
      for (int c = 0; c < C_CH; ++c) acc[c] = 0.0f;
      gs_accum(pXY, R, py, px, acc);   // _grid_sample(xy, py, px)
      gs_accum(pXZ, R, pz, px, acc);   // _grid_sample(xz, pz, px)
      gs_accum(pYZ, R, pz, py, acc);   // _grid_sample(yz, pz, py)
      #pragma unroll
      for (int c = 0; c < C_CH; ++c)
        featW[m * K1P + lvl * C_CH + c] = (_Float16)acc[c];
    }

    // ---- Layer 1: [16 x 49] @ [49 x 168] -> relu -> h1 (epilogue pipelined) ----
    {
      v16h a0 = load_a_frag(featW, K1P, 0, lane);
      v16h a1 = load_a_frag(featW, K1P, 1, lane);
      v8f accP = {};
      #pragma unroll
      for (int nt = 0; nt < 12; ++nt) {
        v8f acc = {};
        v16h bf = load_b_frag(w1s, K1P, 0, nt, lane);
        acc = __builtin_amdgcn_wmma_f32_16x16x32_f16(false, a0, false, bf, (short)0, acc, false, false);
        bf = load_b_frag(w1s, K1P, 1, nt, lane);
        acc = __builtin_amdgcn_wmma_f32_16x16x32_f16(false, a1, false, bf, (short)0, acc, false, false);
        if (nt > 0) store_relu8(h1W, (nt - 1) * 16 + m, mOff, accP);  // drain prev while WMMA in flight
        accP = acc;
      }
      store_relu8(h1W, 11 * 16 + m, mOff, accP);
      if (m == 8) {                       // bias-input column for next layer
        #pragma unroll
        for (int r = 0; r < 8; ++r) h1W[(mOff + r) * NP + MLP_HW] = (_Float16)1.0f;
      }
    }

    // ---- Layer 2: [16 x 169] @ [169 x 168] -> relu -> h2 (epilogue pipelined) ----
    {
      v16h af[6];
      #pragma unroll
      for (int kt = 0; kt < 6; ++kt) af[kt] = load_a_frag(h1W, NP, kt, lane);
      v8f accP = {};
      #pragma unroll
      for (int nt = 0; nt < 12; ++nt) {
        v8f acc = {};
        #pragma unroll
        for (int kt = 0; kt < 6; ++kt) {
          v16h bf = load_b_frag(w2s, NP, kt, nt, lane);
          acc = __builtin_amdgcn_wmma_f32_16x16x32_f16(false, af[kt], false, bf, (short)0, acc, false, false);
        }
        if (nt > 0) store_relu8(h2W, (nt - 1) * 16 + m, mOff, accP);
        accP = acc;
      }
      store_relu8(h2W, 11 * 16 + m, mOff, accP);
      if (m == 8) {
        #pragma unroll
        for (int r = 0; r < 8; ++r) h2W[(mOff + r) * NP + MLP_HW] = (_Float16)1.0f;
      }
    }

    // ---- Layer 3: [16 x 169] @ [169 x 8] -> output transforms ----
    {
      v8f acc = {};
      #pragma unroll
      for (int kt = 0; kt < 6; ++kt) {
        v16h af = load_a_frag(h2W, NP, kt, lane);
        v16h bf = load_b_frag(w3s, NP, kt, 0, lane);
        acc = __builtin_amdgcn_wmma_f32_16x16x32_f16(false, af, false, bf, (short)0, acc, false, false);
      }
      int n = m;                       // output channel this lane holds
      #pragma unroll
      for (int r = 0; r < 8; ++r) {
        float v = acc[r];
        int p = ptBase + mOff + r;
        if (p < NPTS && n < OUT_CH) {
          if (n < 4) {
            __builtin_nontemporal_store(v, &out[4 * NPTS + p * 4 + n]);            // rotations
          } else if (n < 7) {
            float s = 1.0f / (1.0f + __expf(-v));
            __builtin_nontemporal_store(s, &out[NPTS + p * 3 + (n - 4)]);          // scale
          } else {
            __builtin_nontemporal_store(v * 10.0f, &out[p]);                       // opacity*10
          }
        }
      }
    }
    // wave-private LDS regions + in-order DS => no barrier needed per tile
  }
}

extern "C" void kernel_launch(void* const* d_in, const int* in_sizes, int n_in,
                              void* d_out, int out_size, void* d_ws, size_t ws_size,
                              hipStream_t stream) {
  const float* xyz = (const float*)d_in[0];
  const float* xy0 = (const float*)d_in[1];
  const float* xz0 = (const float*)d_in[2];
  const float* yz0 = (const float*)d_in[3];
  const float* xy1 = (const float*)d_in[4];
  const float* xz1 = (const float*)d_in[5];
  const float* yz1 = (const float*)d_in[6];
  const float* W1  = (const float*)d_in[7];
  const float* b1  = (const float*)d_in[8];
  const float* W2  = (const float*)d_in[9];
  const float* b2  = (const float*)d_in[10];
  const float* W3  = (const float*)d_in[11];
  const float* b3  = (const float*)d_in[12];
  float* out = (float*)d_out;

  int nTiles = (NPTS + PTS_PER_BLOCK - 1) / PTS_PER_BLOCK;   // 5209
  int blocks = nTiles < 1536 ? nTiles : 1536;                // grid-stride, amortize weight staging
  triplane_mlp_wmma<<<blocks, THREADS, SMEM_BYTES, stream>>>(
      xyz, xy0, xz0, yz0, xy1, xz1, yz1, W1, b1, W2, b2, W3, b3, out, nTiles);
}
